// MultiHeadAttention_34651796144477
// MI455X (gfx1250) — matrix operand order repaired and checked by reading, not compiled, not run on hardware
//
#include <hip/hip_runtime.h>
#include <hip/hip_bf16.h>

// ---------------------------------------------------------------------------
// Multi-head attention, restructured via the low-rank value path:
//   out = sum_h Wo[h] * softmax(mask(QK^T/8)) @ (x @ Wvd_h) @ Wvu_h
// (~52 GFLOP instead of ~200 GFLOP; all GEMMs on v_wmma_f32_16x16x32_bf16.)
// Attention K/V blocks are staged once per workgroup into double-buffered LDS
// via the CDNA5 async global->LDS engine (ASYNCcnt), overlapping the copy of
// block kb+1 with the WMMA+softmax work on block kb.
// ---------------------------------------------------------------------------

typedef __bf16 bf16_t;
typedef __attribute__((ext_vector_type(16))) __bf16 v16bf;
typedef __attribute__((ext_vector_type(8)))  __bf16 v8bf;
typedef __attribute__((ext_vector_type(2)))  __bf16 v2bf;
typedef __attribute__((ext_vector_type(8)))  float  v8f;
typedef __attribute__((ext_vector_type(4)))  int    v4i;

static constexpr int Bn = 4, Sn = 2048, En = 512, Dn = 64, Hn = 8;

static __device__ __forceinline__ v8f wmma_bf16(v16bf a, v16bf b, v8f c) {
  return __builtin_amdgcn_wmma_f32_16x16x32_bf16(false, a, false, b,
                                                 (short)0, c, false, false);
}
static __device__ __forceinline__ v16bf cat16(v8bf lo, v8bf hi) {
  return __builtin_shufflevector(lo, hi, 0, 1, 2, 3, 4, 5, 6, 7,
                                         8, 9, 10, 11, 12, 13, 14, 15);
}

// --- CDNA5 async global->LDS copy (guarded: falls back to cooperative copy) --
#if __has_builtin(__builtin_amdgcn_global_load_async_to_lds_b128)
#define HAVE_ASYNC_LDS 1
// builtin params are int4* with global/LDS address spaces (per hipcc diag)
#define ASYNC_GPTR(p) ((__attribute__((address_space(1))) v4i*)(p))
#define ASYNC_LPTR(p) ((__attribute__((address_space(3))) v4i*)(p))
#else
#define HAVE_ASYNC_LDS 0
#endif

static __device__ __forceinline__ void wait_async0() {
#if __has_builtin(__builtin_amdgcn_s_wait_asynccnt)
  __builtin_amdgcn_s_wait_asynccnt(0);
#else
  asm volatile("s_wait_asynccnt 0x0" ::: "memory");
#endif
}

// Fragment lane mapping (wave32, per CDNA5 ISA 7.12.2):
//   A (16x32, MxK): m = lane%16, g = lane/16; half i -> k = 8g + i%8 + 16*(i>=8)
//   B (32x16, KxN): n = lane%16, g = lane/16; half i -> same k as A
//   C/D (16x16 f32): n = lane%16, g = lane/16; vgpr r -> m = r + 8g

// ---------------------------------------------------------------------------
// Kernel 1: projections.  grid = (3 [q/k/vd], B*H, S/128), block = 256 (8 waves)
// Q,K stored [S][D]; Vd stored transposed [D][S].
// ---------------------------------------------------------------------------
__global__ __launch_bounds__(256) void proj_kernel(
    const float* __restrict__ x,  const float* __restrict__ Wq,
    const float* __restrict__ Wk, const float* __restrict__ Wvd,
    bf16_t* __restrict__ Qb, bf16_t* __restrict__ Kb, bf16_t* __restrict__ Vt)
{
  constexpr int EC = 128;
  __shared__ bf16_t Wlds[Dn * EC];           // transposed [d][e], 16 KB
  const int proj = blockIdx.x;
  const int bh   = blockIdx.y;
  const int b    = bh / Hn;
  const float* W = (proj == 0 ? Wq : (proj == 1 ? Wk : Wvd)) +
                   (size_t)(bh % Hn) * En * Dn;
  const float ascale = (proj == 0) ? 0.125f : 1.0f;   // fold 1/sqrt(D) into Q

  const int wave = threadIdx.x >> 5;
  const int lane = threadIdx.x & 31;
  const int n = lane & 15;
  const int g = lane >> 4;
  const int s0 = blockIdx.z * 128 + wave * 16;
  const float* xrow = x + ((size_t)b * Sn + (s0 + n)) * En;

  v8f c[4] = {};

  for (int ec = 0; ec < En; ec += EC) {
    __syncthreads();
    for (int i = threadIdx.x; i < EC * Dn; i += 256) {
      const int e = i / Dn, d = i % Dn;
      Wlds[d * EC + e] = (bf16_t)W[(size_t)ec * Dn + i];
    }
    __syncthreads();

    for (int e0 = 0; e0 < EC; e0 += 32) {
      const v8f x0 = *(const v8f*)(xrow + ec + e0 + 16 * g);
      const v8f x1 = *(const v8f*)(xrow + ec + e0 + 16 * g + 8);
      v16bf bb[4];
      #pragma unroll
      for (int j = 0; j < 4; ++j)
        bb[j] = *(const v16bf*)&Wlds[(16 * j + n) * EC + e0 + 16 * g];
      v16bf a;
      #pragma unroll
      for (int i = 0; i < 8; ++i) {
        a[i]     = (bf16_t)(x0[i] * ascale);
        a[i + 8] = (bf16_t)(x1[i] * ascale);
      }
      #pragma unroll
      for (int j = 0; j < 4; ++j) c[j] = wmma_bf16(a, bb[j], c[j]);
    }
  }

  if (proj < 2) {                           // Q/K: row-major [S][D]
    bf16_t* Out = (proj == 0 ? Qb : Kb) + (size_t)bh * Sn * Dn;
    #pragma unroll
    for (int j = 0; j < 4; ++j)
      #pragma unroll
      for (int r = 0; r < 8; ++r)
        Out[(size_t)(s0 + r + 8 * g) * Dn + 16 * j + n] = (bf16_t)c[j][r];
  } else {                                  // Vd: transposed [D][S]
    bf16_t* Out = Vt + (size_t)bh * Sn * Dn;
    #pragma unroll
    for (int j = 0; j < 4; ++j)
      #pragma unroll
      for (int r = 0; r < 8; ++r)
        Out[(size_t)(16 * j + n) * Sn + s0 + r + 8 * g] = (bf16_t)c[j][r];
  }
}

// ---------------------------------------------------------------------------
// Kernel 2: causal flash attention.  grid = (B*H, S/128), block = 256.
// K/V blocks staged once per workgroup into double-buffered LDS (async DMA),
// all 8 waves consume them from LDS.  Two barriers per block iteration:
//   sync#1: buf[cur] fully written (issuer waited ASYNCcnt before it)
//   sync#2: all reads of buf[cur] retired before it is overwritten (kb+2)
// ---------------------------------------------------------------------------
__global__ __launch_bounds__(256) void attn_kernel(
    const bf16_t* __restrict__ Qb, const bf16_t* __restrict__ Kb,
    const bf16_t* __restrict__ Vt, bf16_t* __restrict__ Ob)
{
  __shared__ bf16_t Klds[2][32 * Dn];   // 2 x 4 KB, rows [key][d]
  __shared__ bf16_t Vlds[2][Dn * 32];   // 2 x 4 KB, rows [d][key]
  __shared__ v2bf   pstv[8][16 * 16];   // 8 KB P bounce (packed pairs)

  const int bh   = blockIdx.x;
  const int wave = threadIdx.x >> 5;
  const int lane = threadIdx.x & 31;
  const int n = lane & 15;
  const int g = lane >> 4;
  const int s0 = blockIdx.y * 128 + wave * 16;
  const bf16_t* Qh = Qb + (size_t)bh * Sn * Dn;
  const bf16_t* Kh = Kb + (size_t)bh * Sn * Dn;
  const bf16_t* Vh = Vt + (size_t)bh * Sn * Dn;   // [D][S]

  // ---- stage one 32-key K/V block into LDS buffer `buf` ----
  auto stage = [&](int c0, int buf) {
#if HAVE_ASYNC_LDS
    if (wave == 0) {
      const bf16_t* kg = Kh + (size_t)c0 * Dn;          // 4 KB contiguous
      #pragma unroll
      for (int q = 0; q < 8; ++q)
        __builtin_amdgcn_global_load_async_to_lds_b128(
            ASYNC_GPTR(kg + q * 256 + lane * 8),
            ASYNC_LPTR(&Klds[buf][q * 256 + lane * 8]), 0, 0);
      const int d0 = lane >> 2, ch = lane & 3;          // 64 rows x 64 B
      #pragma unroll
      for (int q = 0; q < 8; ++q)
        __builtin_amdgcn_global_load_async_to_lds_b128(
            ASYNC_GPTR(Vh + (size_t)(q * 8 + d0) * Sn + c0 + ch * 8),
            ASYNC_LPTR(&Vlds[buf][(q * 8 + d0) * 32 + ch * 8]), 0, 0);
    }
#else
    const int t = threadIdx.x;                          // cooperative fallback
    *(uint4*)&Klds[buf][t * 8] = *(const uint4*)(Kh + (size_t)c0 * Dn + t * 8);
    const int d0 = t >> 2, ch = t & 3;
    *(uint4*)&Vlds[buf][d0 * 32 + ch * 8] =
        *(const uint4*)(Vh + (size_t)d0 * Sn + c0 + ch * 8);
#endif
  };

  // Q A-fragments for the whole tile (D=64 -> 2 fragments of K=32)
  v16bf qf[2];
  #pragma unroll
  for (int f = 0; f < 2; ++f) {
    const bf16_t* qp = Qh + (size_t)(s0 + n) * Dn + 32 * f + 16 * g;
    qf[f] = cat16(*(const v8bf*)qp, *(const v8bf*)(qp + 8));
  }

  float mi[8], li[8], fac[8];
  #pragma unroll
  for (int r = 0; r < 8; ++r) { mi[r] = -1e30f; li[r] = 0.0f; }
  v8f acc[4] = {};

  const int nkb    = (s0 + 16 + 31) >> 5;                   // this wave's blocks
  const int nkbmax = (blockIdx.y * 128 + 112 + 16 + 31) >> 5;  // block max (uniform)

  stage(0, 0);                                              // prologue
  for (int kb = 0; kb < nkbmax; ++kb) {
    const int cur = kb & 1;
#if HAVE_ASYNC_LDS
    if (wave == 0) wait_async0();
#endif
    __syncthreads();                       // buf[cur] ready for everyone
    if (kb + 1 < nkbmax) stage((kb + 1) * 32, cur ^ 1);

    if (kb < nkb) {
      const int c0 = kb * 32;
      // ---- scores from LDS: 8x ds_load_b128, 4 back-to-back WMMAs ----
      v16bf kf[2][2];
      #pragma unroll
      for (int t = 0; t < 2; ++t)
        #pragma unroll
        for (int f = 0; f < 2; ++f)
          kf[t][f] = *(const v16bf*)&Klds[cur][(16 * t + n) * Dn + 32 * f + 16 * g];
      v8f sc[2] = {};
      #pragma unroll
      for (int t = 0; t < 2; ++t)
        #pragma unroll
        for (int f = 0; f < 2; ++f) sc[t] = wmma_bf16(qf[f], kf[t][f], sc[t]);

      // ---- online softmax; row stats via 16-lane shuffle reductions ----
      #pragma unroll
      for (int r = 0; r < 8; ++r) {
        const int row = s0 + r + 8 * g;
        float v0 = sc[0][r];
        float v1 = sc[1][r];
        v0 = ((c0 + n)      <= row && v0 != 0.0f) ? v0 : -1e30f; // tril + !=0
        v1 = ((c0 + 16 + n) <= row && v1 != 0.0f) ? v1 : -1e30f;
        float bm = fmaxf(v0, v1);
        #pragma unroll
        for (int off = 8; off > 0; off >>= 1)
          bm = fmaxf(bm, __shfl_xor(bm, off, 32));
        const float mnew = fmaxf(mi[r], bm);
        const float p0 = __expf(v0 - mnew);
        const float p1 = __expf(v1 - mnew);
        float ps = p0 + p1;
        #pragma unroll
        for (int off = 8; off > 0; off >>= 1) ps += __shfl_xor(ps, off, 32);
        fac[r] = __expf(mi[r] - mnew);
        li[r]  = li[r] * fac[r] + ps;
        mi[r]  = mnew;
        v2bf pk;                                   // packed pair -> one b32 store
        pk[0] = (bf16_t)p0;
        pk[1] = (bf16_t)p1;
        pstv[wave][(r + 8 * g) * 16 + n] = pk;
      }
      #pragma unroll
      for (int j = 0; j < 4; ++j)
        #pragma unroll
        for (int r = 0; r < 8; ++r) acc[j][r] *= fac[r];

      // ---- P back in A-layout: 2x ds_load_b128 + register de-interleave ----
      const v16bf raw = *(const v16bf*)&pstv[wave][n * 16 + 8 * g];
      const v16bf pa = __builtin_shufflevector(raw, raw,
          0, 2, 4, 6, 8, 10, 12, 14, 1, 3, 5, 7, 9, 11, 13, 15);

      // ---- acc += P @ Vd from LDS: 4 back-to-back WMMAs ----
      v16bf vf[4];
      #pragma unroll
      for (int j = 0; j < 4; ++j)
        vf[j] = cat16(*(const v8bf*)&Vlds[cur][(16 * j + n) * 32 + 8 * g],
                      *(const v8bf*)&Vlds[cur][(16 * j + n) * 32 + 16 + 8 * g]);
      #pragma unroll
      for (int j = 0; j < 4; ++j) acc[j] = wmma_bf16(pa, vf[j], acc[j]);
    }
    __syncthreads();                       // reads of buf[cur] retired
  }

  bf16_t* Oh = Ob + (size_t)bh * Sn * Dn;
  #pragma unroll
  for (int j = 0; j < 4; ++j)
    #pragma unroll
    for (int r = 0; r < 8; ++r)
      Oh[(size_t)(s0 + r + 8 * g) * Dn + 16 * j + n] = (bf16_t)(acc[j][r] / li[r]);
}

// ---------------------------------------------------------------------------
// Kernel 3a: one-time weight prep: Wt[h][e][d] = bf16(Wvu[h][d][e] * Wo[h])
// ---------------------------------------------------------------------------
__global__ __launch_bounds__(256) void wprep_kernel(
    const float* __restrict__ Wvu, const float* __restrict__ Wo,
    bf16_t* __restrict__ Wt)
{
  const int idx = blockIdx.x * 256 + threadIdx.x;     // H*D*E total
  const int h = idx / (Dn * En);
  const int r = idx % (Dn * En);
  const int d = r / En;
  const int e = r % En;
  Wt[((size_t)h * En + e) * Dn + d] = (bf16_t)(Wvu[idx] * Wo[h]);
}

// ---------------------------------------------------------------------------
// Kernel 3b: up-projection + weighted head sum (Wo folded into Wt).
// ---------------------------------------------------------------------------
__global__ __launch_bounds__(256) void outproj_kernel(
    const bf16_t* __restrict__ Ob, const bf16_t* __restrict__ Wt,
    float* __restrict__ out)
{
  const int wave = threadIdx.x >> 5;
  const int lane = threadIdx.x & 31;
  const int n = lane & 15;
  const int g = lane >> 4;
  const int tile = blockIdx.x * 8 + wave;          // B * (S/16) * (E/16) tiles
  const int b  = tile / (128 * 32);
  const int t  = tile % (128 * 32);
  const int s0 = (t >> 5) * 16;
  const int e0 = (t & 31) * 16;

  v8f c = {};
  #pragma unroll 2                                 // overlap next head's loads
  for (int h = 0; h < Hn; ++h) {
    const bf16_t* Oh = Ob + (size_t)(b * Hn + h) * Sn * Dn;
    const bf16_t* Wh = Wt + (size_t)h * En * Dn;
    v16bf a[2], bb[2];
    #pragma unroll
    for (int f = 0; f < 2; ++f) {
      const bf16_t* op = Oh + (size_t)(s0 + n) * Dn + 32 * f + 16 * g;
      a[f]  = cat16(*(const v8bf*)op, *(const v8bf*)(op + 8));
      bb[f] = *(const v16bf*)(Wh + (size_t)(e0 + n) * Dn + 32 * f + 16 * g);
    }
    #pragma unroll
    for (int f = 0; f < 2; ++f) c = wmma_bf16(a[f], bb[f], c);
  }
  #pragma unroll
  for (int r = 0; r < 8; ++r)
    out[((size_t)b * Sn + s0 + r + 8 * g) * En + e0 + n] = c[r];
}

// ---------------------------------------------------------------------------
extern "C" void kernel_launch(void* const* d_in, const int* in_sizes, int n_in,
                              void* d_out, int out_size, void* d_ws, size_t ws_size,
                              hipStream_t stream) {
  (void)in_sizes; (void)n_in; (void)out_size; (void)ws_size;
  const float* x   = (const float*)d_in[0];
  const float* Wq  = (const float*)d_in[1];
  const float* Wk  = (const float*)d_in[2];
  const float* Wvd = (const float*)d_in[3];
  const float* Wvu = (const float*)d_in[4];
  const float* Wo  = (const float*)d_in[5];
  float* out = (float*)d_out;

  const size_t per = (size_t)Bn * Hn * Sn * Dn;   // 4.19M elems per tensor
  bf16_t* Qb = (bf16_t*)d_ws;                      // ws use: 4*per*2 + 512KB
  bf16_t* Kb = Qb + per;
  bf16_t* Vt = Kb + per;
  bf16_t* Oh = Vt + per;
  bf16_t* Wt = Oh + per;                           // H*E*D bf16 = 512 KB

  wprep_kernel<<<dim3(Hn * Dn * En / 256), 256, 0, stream>>>(Wvu, Wo, Wt);
  proj_kernel<<<dim3(3, Bn * Hn, Sn / 128), 256, 0, stream>>>(x, Wq, Wk, Wvd,
                                                              Qb, Kb, Vt);
  attn_kernel<<<dim3(Bn * Hn, Sn / 128), 256, 0, stream>>>(Qb, Kb, Vt, Oh);
  outproj_kernel<<<dim3(Bn * 128 * 32 / 8), 256, 0, stream>>>(Oh, Wt, out);
}
